// RGPRGNN_10900626997774
// MI455X (gfx1250) — matrix operand order repaired and checked by reading, not compiled, not run on hardware
//
#include <hip/hip_runtime.h>

typedef __attribute__((ext_vector_type(16))) _Float16 v16h;
typedef __attribute__((ext_vector_type(8)))  float    v8f;

#define NNODES 50000
#define NEDGES 800000
#define INC    256
#define HIDC   128
#define OUTC   2
#define NREL   8
#define NBASE  4
#define NLAYER 3

// ---------------- elementwise helpers ----------------

__global__ void k_f32_to_f16(const float* __restrict__ in, _Float16* __restrict__ out, int n) {
    int i = blockIdx.x * blockDim.x + threadIdx.x;
    if (i < n) out[i] = (_Float16)in[i];
}

__global__ void k_zero_f32(float* __restrict__ p, int n) {
    int i = blockIdx.x * blockDim.x + threadIdx.x;
    if (i < n) p[i] = 0.0f;
}

// hidden = temp[idx] * cur
__global__ void k_scale(const float* __restrict__ in, float* __restrict__ out,
                        const float* __restrict__ temp, int idx, int n) {
    int i = blockIdx.x * blockDim.x + threadIdx.x;
    if (i < n) out[i] = temp[idx] * in[i];
}

// cur = relu?(cur); hidden += temp[idx] * cur
__global__ void k_relu_axpy(float* __restrict__ cur, float* __restrict__ hidden,
                            const float* __restrict__ temp, int idx, int do_relu, int n) {
    int i = blockIdx.x * blockDim.x + threadIdx.x;
    if (i < n) {
        float c = cur[i];
        if (do_relu) c = fmaxf(c, 0.0f);
        cur[i] = c;
        hidden[i] += temp[idx] * c;
    }
}

// ---------------- weight panel builders (B^T, f16) ----------------

// Bt[j][k] = lin1_w[k][j]   (j<128 out cols, k<256 in dim)
__global__ void k_build_bt_lin1(const float* __restrict__ w, _Float16* __restrict__ bt) {
    int i = blockIdx.x * blockDim.x + threadIdx.x;
    if (i >= HIDC * INC) return;
    int j = i / INC, k = i - j * INC;
    bt[i] = (_Float16)w[k * HIDC + j];
}

// rows 0..127: root_l^T ; rows 128..1151: W_r^T with W_r = sum_b comp[l,r,b]*bases[l,b]
__global__ void k_build_bt_layer(const float* __restrict__ root, const float* __restrict__ bases,
                                 const float* __restrict__ comp, _Float16* __restrict__ bt, int l) {
    int i = blockIdx.x * blockDim.x + threadIdx.x;
    if (i >= (HIDC + NREL * HIDC) * HIDC) return;
    int j = i / HIDC, k = i - j * HIDC;
    float v;
    if (j < HIDC) {
        v = root[(l * HIDC + k) * HIDC + j];
    } else {
        int r = (j - HIDC) >> 7;
        int o = (j - HIDC) & 127;
        const float* cb = comp + (l * NREL + r) * NBASE;
        v = 0.0f;
#pragma unroll
        for (int b = 0; b < NBASE; ++b)
            v += cb[b] * bases[((l * NBASE + b) * HIDC + k) * HIDC + o];
    }
    bt[i] = (_Float16)v;
}

// ---------------- WMMA GEMM: C[M,Nc] = A[M,K](f16) * Bt[Nc,K]^T(f16), f32 acc ----------------
// One wave computes a 16(M) x 128(N) tile: 1 A-fragment reused over 8 wmma per K-step of 32.
// K is compile-time so the k-loop fully unrolls and loads pipeline across WMMAs.
// Requires M%16==0, K%32==0, Nc%128==0 (true for all call sites; M=50000=3125*16).

template <int K>
__global__ void k_wmma_gemm(const _Float16* __restrict__ A, const _Float16* __restrict__ Bt,
                            float* __restrict__ C, const float* __restrict__ bias,
                            int M, int Nc, int do_relu) {
    const int NT = 8;                                  // 8 col-tiles of 16 => 128 cols/wave
    int wave = blockIdx.x * (blockDim.x >> 5) + (threadIdx.x >> 5);
    int NG = Nc >> 7;                                  // 128-wide N groups
    int mt = wave / NG;
    int ng = wave - mt * NG;
    if (mt >= (M >> 4)) return;                        // wave-uniform: EXEC stays all-ones below

    int lane = threadIdx.x & 31;
    int half = lane >> 4;                              // 0: lanes 0-15, 1: lanes 16-31
    int lm   = lane & 15;

    // A fragment base: row = mt*16+lm; K groups {k0+half*8 .. +7} and {k0+16+half*8 .. +7}
    const _Float16* arow = A + (size_t)((mt << 4) + lm) * K + half * 8;
    // B fragment base: column ng*128+t*16+lm; K groups {k0+half*8 .. +7} and {k0+16+half*8 .. +7}
    const _Float16* bbase = Bt + (size_t)((ng << 7) + lm) * K + half * 8;

    v8f acc[NT];
#pragma unroll
    for (int t = 0; t < NT; ++t)
#pragma unroll
        for (int e = 0; e < 8; ++e) acc[t][e] = 0.0f;

#pragma unroll
    for (int k0 = 0; k0 < K; k0 += 32) {
        union { v16h v; uint4 u[2]; } af;
        af.u[0] = *(const uint4*)(arow + k0);
        af.u[1] = *(const uint4*)(arow + k0 + 16);
#pragma unroll
        for (int t = 0; t < NT; ++t) {
            const _Float16* brow = bbase + (size_t)(t << 4) * K + k0;
            union { v16h v; uint4 u[2]; } bf;
            bf.u[0] = *(const uint4*)(brow);
            bf.u[1] = *(const uint4*)(brow + 16);
            acc[t] = __builtin_amdgcn_wmma_f32_16x16x32_f16(
                false, af.v, false, bf.v, (short)0, acc[t], false, false);
        }
    }

    // C/D layout: VGPR v holds row mt*16 + half*8 + v, col = n0 + lm
#pragma unroll
    for (int t = 0; t < NT; ++t) {
        int col = (ng << 7) + (t << 4) + lm;
        float bv = bias ? bias[col] : 0.0f;
#pragma unroll
        for (int v = 0; v < 8; ++v) {
            int row = (mt << 4) + (half << 3) + v;
            float val = acc[t][v] + bv;
            if (do_relu) val = fmaxf(val, 0.0f);
            C[(size_t)row * Nc + col] = val;
        }
    }
}

// ---------------- graph kernels ----------------

// cnt[r][dst] += 1 per edge
__global__ void k_count(const int* __restrict__ ei, const int* __restrict__ et,
                        float* __restrict__ cnt, int E, int N) {
    int e = blockIdx.x * blockDim.x + threadIdx.x;
    if (e < E) atomicAdd(&cnt[et[e] * N + ei[E + e]], 1.0f);
}

__global__ void k_invcnt(float* __restrict__ c, int n) {
    int i = blockIdx.x * blockDim.x + threadIdx.x;
    if (i < n) c[i] = 1.0f / fmaxf(c[i], 1.0f);
}

// One wave per edge: out[dst] += hr[src, r*128 : r*128+128] * inv[r][dst]
__global__ void k_scatter(const int* __restrict__ ei, const int* __restrict__ et,
                          const float* __restrict__ hr, const float* __restrict__ inv,
                          float* __restrict__ out, int E, int N) {
    int gid  = blockIdx.x * blockDim.x + threadIdx.x;
    int e    = gid >> 5;
    int lane = gid & 31;
    if (e >= E) return;
    int src = ei[e];
    int dst = ei[E + e];
    int r   = et[e];
    float s = inv[r * N + dst];
    const float4 m = *(const float4*)(hr + (size_t)src * (NREL * HIDC) + r * HIDC + lane * 4);
    float* o = out + (size_t)dst * HIDC + lane * 4;
    atomicAdd(o + 0, m.x * s);
    atomicAdd(o + 1, m.y * s);
    atomicAdd(o + 2, m.z * s);
    atomicAdd(o + 3, m.w * s);
}

// One wave per node: out[n, 0:2] = hidden[n,:] @ lin2_w + lin2_b
__global__ void k_final(const float* __restrict__ h, const float* __restrict__ w,
                        const float* __restrict__ b, float* __restrict__ out, int N) {
    int gid  = blockIdx.x * blockDim.x + threadIdx.x;
    int node = gid >> 5;
    int lane = gid & 31;
    if (node >= N) return;
    const float4 hv = *(const float4*)(h + (size_t)node * HIDC + lane * 4);
    int c = lane * 4;
    float s0 = hv.x * w[(c + 0) * OUTC] + hv.y * w[(c + 1) * OUTC] +
               hv.z * w[(c + 2) * OUTC] + hv.w * w[(c + 3) * OUTC];
    float s1 = hv.x * w[(c + 0) * OUTC + 1] + hv.y * w[(c + 1) * OUTC + 1] +
               hv.z * w[(c + 2) * OUTC + 1] + hv.w * w[(c + 3) * OUTC + 1];
#pragma unroll
    for (int off = 16; off > 0; off >>= 1) {
        s0 += __shfl_down(s0, off, 32);
        s1 += __shfl_down(s1, off, 32);
    }
    if (lane == 0) {
        out[node * OUTC + 0] = s0 + b[0];
        out[node * OUTC + 1] = s1 + b[1];
    }
}

// ---------------- driver ----------------

extern "C" void kernel_launch(void* const* d_in, const int* in_sizes, int n_in,
                              void* d_out, int out_size, void* d_ws, size_t ws_size,
                              hipStream_t stream) {
    const float* x        = (const float*)d_in[0];
    const int*   ei       = (const int*)  d_in[1];   // [2,E]: row0 src, row1 dst
    const int*   et       = (const int*)  d_in[2];   // [E]
    const float* lin1_w   = (const float*)d_in[3];
    const float* lin1_b   = (const float*)d_in[4];
    const float* bases    = (const float*)d_in[5];
    const float* comp     = (const float*)d_in[6];
    const float* root     = (const float*)d_in[7];
    const float* bias_cv  = (const float*)d_in[8];
    const float* lin2_w   = (const float*)d_in[9];
    const float* lin2_b   = (const float*)d_in[10];
    const float* temp     = (const float*)d_in[11];
    float* out = (float*)d_out;
    (void)in_sizes; (void)n_in; (void)out_size; (void)ws_size;

    // workspace carve-up (256B aligned)
    char* ws = (char*)d_ws;
    size_t off = 0;
    auto carve = [&](size_t bytes) -> char* {
        char* p = ws + off;
        off = (off + bytes + 255) & ~(size_t)255;
        return p;
    };
    _Float16* x16    = (_Float16*)carve((size_t)NNODES * INC  * sizeof(_Float16)); // 25.6 MB
    _Float16* cur16  = (_Float16*)carve((size_t)NNODES * HIDC * sizeof(_Float16)); // 12.8 MB
    float*    cur    = (float*)   carve((size_t)NNODES * HIDC * sizeof(float));    // 25.6 MB
    float*    hidden = (float*)   carve((size_t)NNODES * HIDC * sizeof(float));    // 25.6 MB
    float*    inv    = (float*)   carve((size_t)NREL * NNODES * sizeof(float));    //  1.6 MB
    _Float16* bt     = (_Float16*)carve((size_t)(HIDC + NREL * HIDC) * HIDC * sizeof(_Float16));
    float*    hr     = (float*)   carve((size_t)NNODES * NREL * HIDC * sizeof(float)); // 204.8 MB

    const int T = 256;
    const int nh = NNODES * HIDC;
    const int MT = NNODES / 16;                        // 3125 row tiles

    // 1) x -> f16, lin1 weight panel
    k_f32_to_f16<<<(NNODES * INC + T - 1) / T, T, 0, stream>>>(x, x16, NNODES * INC);
    k_build_bt_lin1<<<(HIDC * INC + T - 1) / T, T, 0, stream>>>(lin1_w, bt);

    // 2) cur = relu(x @ lin1_w + lin1_b)  — WMMA GEMM, M=50000 K=256 Nc=128
    {
        int waves  = MT * (HIDC / 128);                // 3125
        int blocks = (waves + 7) / 8;                  // 8 waves / 256-thread block
        k_wmma_gemm<INC><<<blocks, T, 0, stream>>>(x16, bt, cur, lin1_b, NNODES, HIDC, 1);
    }

    // 3) hidden = temp[0] * cur
    k_scale<<<(nh + T - 1) / T, T, 0, stream>>>(cur, hidden, temp, 0, nh);

    // 4) per-(relation,dst) mean normalizers (layer-invariant)
    k_zero_f32<<<(NREL * NNODES + T - 1) / T, T, 0, stream>>>(inv, NREL * NNODES);
    k_count<<<(NEDGES + T - 1) / T, T, 0, stream>>>(ei, et, inv, NEDGES, NNODES);
    k_invcnt<<<(NREL * NNODES + T - 1) / T, T, 0, stream>>>(inv, NREL * NNODES);

    // 5) layers
    for (int l = 0; l < NLAYER; ++l) {
        k_f32_to_f16<<<(nh + T - 1) / T, T, 0, stream>>>(cur, cur16, nh);
        k_build_bt_layer<<<((HIDC + NREL * HIDC) * HIDC + T - 1) / T, T, 0, stream>>>(
            root, bases, comp, bt, l);

        // root/self transform + bias, overwrites cur (cur16 holds the layer input)
        {
            int waves  = MT * (HIDC / 128);            // 3125
            int blocks = (waves + 7) / 8;
            k_wmma_gemm<HIDC><<<blocks, T, 0, stream>>>(cur16, bt, cur, bias_cv + l * HIDC,
                                                        NNODES, HIDC, 0);
        }
        // all 8 relation transforms fused: hr[N, 1024]
        {
            int waves  = MT * ((NREL * HIDC) / 128);   // 3125 * 8
            int blocks = (waves + 7) / 8;
            k_wmma_gemm<HIDC><<<blocks, T, 0, stream>>>(cur16, bt + HIDC * HIDC, hr, nullptr,
                                                        NNODES, NREL * HIDC, 0);
        }
        // edge scatter with folded mean normalization
        k_scatter<<<((size_t)NEDGES * 32 + T - 1) / T, T, 0, stream>>>(
            ei, et, hr, inv, cur, NEDGES, NNODES);

        // cur = relu?(cur); hidden += temp[l+1]*cur
        k_relu_axpy<<<(nh + T - 1) / T, T, 0, stream>>>(cur, hidden, temp, l + 1,
                                                        (l < NLAYER - 1) ? 1 : 0, nh);
    }

    // 6) out = hidden @ lin2_w + lin2_b
    k_final<<<(NNODES * 32 + T - 1) / T, T, 0, stream>>>(hidden, lin2_w, lin2_b, out, NNODES);
}
